// SimpleLinear_86294482911649
// MI455X (gfx1250) — compile-verified, matrix operand order checked by reading, not run on hardware
//
#include <hip/hip_runtime.h>
#include <stdint.h>

// Problem dims (fixed by the reference)
constexpr int  BDIM = 8192;
constexpr int  INF  = 4096;
constexpr int  OUTF = 4096;
constexpr int  G    = 8;
constexpr float EPSV = 1e-5f;
constexpr float QBV  = 128.0f;

typedef __attribute__((ext_vector_type(16))) _Float16 v16h;
typedef __attribute__((ext_vector_type(8)))  float    v8f;
typedef __attribute__((ext_vector_type(4)))  int      v4i;

// pointer types demanded by __builtin_amdgcn_global_load_async_to_lds_b128
// (diagnostic showed param0 = v4i in addrspace(1); LDS side is addrspace(3))
typedef __attribute__((address_space(1))) v4i* gas_v4i_p;
typedef __attribute__((address_space(3))) v4i* las_v4i_p;

union F16x16 {
    _Float16 h[16];
    uint4    q[2];
    v16h     v;
};

// CDNA5 async global->LDS copy (ASYNCcnt-tracked), guarded so compilation
// cannot regress on toolchains that lack the builtin.
#if defined(__has_builtin)
#if __has_builtin(__builtin_amdgcn_global_load_async_to_lds_b128)
#define HAVE_ASYNC_LDS 1
#endif
#endif

__device__ __forceinline__ void wait_asynccnt0() {
#if defined(__has_builtin)
#if __has_builtin(__builtin_amdgcn_s_wait_asynccnt)
    __builtin_amdgcn_s_wait_asynccnt(0);
    return;
#else
    asm volatile("s_wait_asynccnt 0x0" ::: "memory");
    return;
#endif
#else
    asm volatile("s_wait_asynccnt 0x0" ::: "memory");
#endif
}

// ---------------------------------------------------------------------------
// 0) zero the small reduction slots in ws
// ---------------------------------------------------------------------------
__global__ void k_init(float* __restrict__ sums, unsigned* __restrict__ gbits) {
    int t = threadIdx.x;
    if (t < G) { sums[t] = 0.0f; gbits[t] = 0u; }
}

// ---------------------------------------------------------------------------
// 1) per weight-group sum of |qw|  (group = 512 consecutive output rows)
// ---------------------------------------------------------------------------
__global__ __launch_bounds__(256) void k_rowabs(const float* __restrict__ qw,
                                                float* __restrict__ sums) {
    __shared__ float red[256];
    const int row = blockIdx.x;
    const int t   = threadIdx.x;
    const float* rp = qw + (size_t)row * INF;
    float s = 0.0f;
    for (int j = t; j < INF; j += 256) s += fabsf(rp[j]);
    red[t] = s;
    __syncthreads();
    for (int off = 128; off > 0; off >>= 1) {
        if (t < off) red[t] += red[t + off];
        __syncthreads();
    }
    if (t == 0) atomicAdd(&sums[row >> 9], red[0]);   // 512 rows per group
}

// ---------------------------------------------------------------------------
// 2) bw = sign(qw*alpha - alpha_g), stored as f16 (exact for {-1,0,+1})
// ---------------------------------------------------------------------------
__global__ __launch_bounds__(256) void k_binarize(const float* __restrict__ qw,
                                                  const float* __restrict__ sums,
                                                  _Float16* __restrict__ bwh) {
    const size_t i4 = ((size_t)blockIdx.x * 256 + threadIdx.x) * 4;
    float tot = 0.0f;
    #pragma unroll
    for (int g = 0; g < G; ++g) tot += sums[g];
    const float alpha = tot / (float)((size_t)OUTF * INF);
    const int   row   = (int)(i4 / INF);
    const int   g     = row >> 9;
    const float ag    = sums[g] * alpha / ((float)(OUTF / G) * (float)INF);

    float4 qv = *(const float4*)(qw + i4);
    union { _Float16 h[4]; uint2 u; } o;
    float z;
    z = qv.x * alpha - ag; o.h[0] = (_Float16)((z > 0.f) ? 1.f : ((z < 0.f) ? -1.f : 0.f));
    z = qv.y * alpha - ag; o.h[1] = (_Float16)((z > 0.f) ? 1.f : ((z < 0.f) ? -1.f : 0.f));
    z = qv.z * alpha - ag; o.h[2] = (_Float16)((z > 0.f) ? 1.f : ((z < 0.f) ? -1.f : 0.f));
    z = qv.w * alpha - ag; o.h[3] = (_Float16)((z > 0.f) ? 1.f : ((z < 0.f) ? -1.f : 0.f));
    *(uint2*)(bwh + i4) = o.u;
}

// ---------------------------------------------------------------------------
// 3) GEMM: out[B,OUT] = x[B,IN] (f32, cvt->f16 on the fly) @ bwh[OUT,IN]^T + bias
//    128x128 tile per workgroup, 8 waves (wave32), wave tile 64x32.
//    K-stage 64 = two K=32 WMMA sub-steps per LDS stage (16 WMMAs/stage).
//    W tile staged via GLOBAL_LOAD_ASYNC_TO_LDS when available.
// ---------------------------------------------------------------------------
#define TKS        64   // halves staged per K stage
#define LDS_STRIDE 72   // 64 + 8 halves pad -> conflict-free b128 LDS access

__global__ __launch_bounds__(256) void k_gemm(const float* __restrict__ x,
                                              const _Float16* __restrict__ bwh,
                                              const float* __restrict__ bias,
                                              float* __restrict__ out,
                                              unsigned* __restrict__ gbits) {
    __shared__ __align__(16) _Float16 Xs[128 * LDS_STRIDE];
    __shared__ __align__(16) _Float16 Ws[128 * LDS_STRIDE];
    __shared__ float red[256];

    const int t    = threadIdx.x;
    const int lane = t & 31;
    const int wave = t >> 5;

    const int row0 = blockIdx.y * 128;   // batch rows
    const int col0 = blockIdx.x * 128;   // output cols

    // wave macro-tile: 2 (M) x 4 (N) waves -> 64x32 each
    const int wm = (wave & 1) * 64;
    const int wn = (wave >> 1) * 32;

    v8f acc[4][2] = {};

    // LDS-fill assignment: thread -> (row = t/2, half-base = (t&1)*32)
    const int frow = t >> 1;
    const int fcol = (t & 1) * 32;
    const float*    xrow = x   + (size_t)(row0 + frow) * INF + fcol;
    const _Float16* wrow = bwh + (size_t)(col0 + frow) * INF + fcol;
    const int ldsoff = frow * LDS_STRIDE + fcol;

    // fragment addressing (ISA 7.12.2, 16-bit 16x32 layout):
    // lane L holds matrix row L%16; half-wave selects K-chunks {0..7,16..23} / {8..15,24..31}
    const int mrow = lane & 15;
    const int kc0  = (lane >> 4) * 8;

    for (int k0 = 0; k0 < INF; k0 += TKS) {
        // --- stage X tile (f32 -> f16 convert), 32 halves / thread ---
        const float4* xp = (const float4*)(xrow + k0);
        #pragma unroll
        for (int h = 0; h < 2; ++h) {
            F16x16 xf;
            #pragma unroll
            for (int j = 0; j < 4; ++j) {
                float4 f = xp[4 * h + j];
                xf.h[4 * j + 0] = (_Float16)f.x;
                xf.h[4 * j + 1] = (_Float16)f.y;
                xf.h[4 * j + 2] = (_Float16)f.z;
                xf.h[4 * j + 3] = (_Float16)f.w;
            }
            *(uint4*)&Xs[ldsoff + 16 * h]     = xf.q[0];
            *(uint4*)&Xs[ldsoff + 16 * h + 8] = xf.q[1];
        }

        // --- stage W tile (already f16), 64 bytes / thread ---
#if HAVE_ASYNC_LDS
        #pragma unroll
        for (int j = 0; j < 4; ++j)
            __builtin_amdgcn_global_load_async_to_lds_b128(
                (gas_v4i_p)(wrow + k0 + 8 * j),
                (las_v4i_p)&Ws[ldsoff + 8 * j],
                0, 0);
#else
        {
            const uint4* wp = (const uint4*)(wrow + k0);
            #pragma unroll
            for (int j = 0; j < 4; ++j)
                *(uint4*)&Ws[ldsoff + 8 * j] = wp[j];
        }
#endif

        // speculative prefetch of next K stage (global_prefetch_b8)
        __builtin_prefetch(xrow + k0 + TKS, 0, 1);

#if HAVE_ASYNC_LDS
        wait_asynccnt0();       // own async copies landed in LDS
#endif
        __syncthreads();        // all waves' tiles visible

        // --- two K=32 WMMA sub-steps per stage ---
        #pragma unroll
        for (int kk = 0; kk < TKS; kk += 32) {
            F16x16 af[4], bf[2];
            #pragma unroll
            for (int m = 0; m < 4; ++m) {
                const int base = (wm + m * 16 + mrow) * LDS_STRIDE + kk + kc0;
                af[m].q[0] = *(const uint4*)&Xs[base];
                af[m].q[1] = *(const uint4*)&Xs[base + 16];
            }
            #pragma unroll
            for (int n = 0; n < 2; ++n) {
                const int base = (wn + n * 16 + mrow) * LDS_STRIDE + kk + kc0;
                bf[n].q[0] = *(const uint4*)&Ws[base];
                bf[n].q[1] = *(const uint4*)&Ws[base + 16];
            }
            #pragma unroll
            for (int m = 0; m < 4; ++m)
                #pragma unroll
                for (int n = 0; n < 2; ++n)
                    acc[m][n] = __builtin_amdgcn_wmma_f32_16x16x32_f16(
                        false, af[m].v, false, bf[n].v, (short)0, acc[m][n],
                        false, false);
        }

        __syncthreads();
    }

    // --- epilogue: +bias, store raw f32 out, track max|out| ---
    float amax = 0.0f;
    #pragma unroll
    for (int n = 0; n < 2; ++n) {
        const int colg = col0 + wn + n * 16 + (lane & 15);
        const float bs = bias[colg];
        #pragma unroll
        for (int m = 0; m < 4; ++m) {
            #pragma unroll
            for (int r = 0; r < 8; ++r) {
                const int rowg = row0 + wm + m * 16 + r + ((lane >> 4) << 3);
                const float v  = acc[m][n][r] + bs;
                out[(size_t)rowg * OUTF + colg] = v;
                amax = fmaxf(amax, fabsf(v));
            }
        }
    }

    red[t] = amax;
    __syncthreads();
    for (int off = 128; off > 0; off >>= 1) {
        if (t < off) red[t] = fmaxf(red[t], red[t + off]);
        __syncthreads();
    }
    // this 128-row tile lies fully inside one 1024-row batch group
    if (t == 0) atomicMax(&gbits[row0 >> 10], __float_as_uint(red[0]));
}

// ---------------------------------------------------------------------------
// 4) in-place quantize: q = clip(out*128/(gamma+eps), -128+eps, 128-eps)
// ---------------------------------------------------------------------------
__global__ __launch_bounds__(256) void k_scale(float* __restrict__ out,
                                               const unsigned* __restrict__ gbits) {
    const size_t i4  = ((size_t)blockIdx.x * 256 + threadIdx.x) * 4;
    const int    row = (int)(i4 / OUTF);
    const int    g   = row >> 10;                 // 1024 batch rows per group
    const float gamma = __uint_as_float(gbits[g]);
    const float s  = QBV / (gamma + EPSV);
    const float lo = -QBV + EPSV, hi = QBV - EPSV;
    float4 v = *(float4*)(out + i4);
    v.x = fminf(fmaxf(v.x * s, lo), hi);
    v.y = fminf(fmaxf(v.y * s, lo), hi);
    v.z = fminf(fmaxf(v.z * s, lo), hi);
    v.w = fminf(fmaxf(v.w * s, lo), hi);
    *(float4*)(out + i4) = v;
}

// ---------------------------------------------------------------------------
extern "C" void kernel_launch(void* const* d_in, const int* in_sizes, int n_in,
                              void* d_out, int out_size, void* d_ws, size_t ws_size,
                              hipStream_t stream) {
    (void)in_sizes; (void)n_in; (void)out_size; (void)ws_size;

    const float* x    = (const float*)d_in[0];   // [8192, 4096]
    const float* qw   = (const float*)d_in[1];   // [4096, 4096]
    const float* bias = (const float*)d_in[2];   // [4096]
    float*       out  = (float*)d_out;           // [8192, 4096]

    // ws layout: 8 f32 group |qw| sums | 8 u32 gamma bits | pad | bwh f16 (32 MB)
    float*    sums  = (float*)d_ws;
    unsigned* gbits = (unsigned*)d_ws + G;
    _Float16* bwh   = (_Float16*)((char*)d_ws + 128);

    k_init<<<1, 64, 0, stream>>>(sums, gbits);
    k_rowabs<<<OUTF, 256, 0, stream>>>(qw, sums);

    const int binBlocks = (int)(((size_t)OUTF * INF / 4) / 256);   // 16384
    k_binarize<<<binBlocks, 256, 0, stream>>>(qw, sums, bwh);

    dim3 grid(OUTF / 128, BDIM / 128);                              // 32 x 64
    k_gemm<<<grid, 256, 0, stream>>>(x, bwh, bias, out, gbits);

    const int scaleBlocks = (int)(((size_t)BDIM * OUTF / 4) / 256); // 32768
    k_scale<<<scaleBlocks, 256, 0, stream>>>(out, gbits);
}
